// SwitchMLP_85237920956513
// MI455X (gfx1250) — compile-verified, matrix operand order checked by reading, not run on hardware
//
#include <hip/hip_runtime.h>
#include <math.h>

// ---------------------------------------------------------------------------
// SwitchMLP (dense MoE, top-p router) for MI455X / gfx1250.
// Compute-bound (~3300 FLOP/byte) -> bf16 WMMA (v_wmma_f32_16x16x32_bf16).
// A-tiles staged with async global->LDS (ASYNCcnt) + double-buffered LDS,
// one barrier per K-step; B-tiles converted f32->bf16 through VGPRs (the
// convert VALU co-executes with the WMMA/XDL pipe).
// Workspace layout (needs ~96.3 MiB):
//   [0, 32MiB)              x in bf16            (T*H)
//   [32MiB, +256KiB)        router weights f32   (T*E)
//   [32MiB+256KiB, +64MiB)  mid = silu(g)*u bf16 (T*I)
// ---------------------------------------------------------------------------

#define T_TOK 8192   // S*B
#define H_DIM 2048
#define I_DIM 4096
#define E_NUM 8
#define TOPP  0.9f

#define BM  128      // block tile M
#define BN  64       // block tile N
#define BK  32       // K step (WMMA K)
#define LDT 40       // padded LDS row stride (elements) -> 80B, 16B aligned

typedef __attribute__((ext_vector_type(16))) __bf16 v16bf;
typedef __attribute__((ext_vector_type(8)))  float  v8f;

union FragU { v16bf f; uint4 q[2]; };

__device__ __forceinline__ unsigned short f2bf_rne(float f) {
  union { float f; unsigned int u; } v; v.f = f;
  unsigned int u = v.u;
  u += 0x7FFFu + ((u >> 16) & 1u);   // round to nearest even
  return (unsigned short)(u >> 16);
}
__device__ __forceinline__ unsigned int packbf2(float lo, float hi) {
  return (unsigned int)f2bf_rne(lo) | ((unsigned int)f2bf_rne(hi) << 16);
}

// async copy 16B global -> LDS (per lane); tracked by ASYNCcnt
__device__ __forceinline__ void async_ld_b128(unsigned lds_off,
                                              const void* gaddr) {
  asm volatile("global_load_async_to_lds_b128 %0, %1, off"
               :: "v"(lds_off), "v"(gaddr) : "memory");
}
__device__ __forceinline__ void wait_async0() {
  asm volatile("s_wait_asynccnt 0x0" ::: "memory");
}
// generic (shared) pointer low 32 bits == LDS byte offset (flat aperture rule)
__device__ __forceinline__ unsigned lds_off32(const void* p) {
  return (unsigned)(unsigned long long)p;
}

// --------------------------- x : f32 -> bf16 -------------------------------
__global__ void cvt_x_kernel(const float* __restrict__ x,
                             unsigned short* __restrict__ xb) {
  size_t i = (size_t)blockIdx.x * blockDim.x + threadIdx.x;   // 8 elems each
  const float4* s = (const float4*)x + i * 2;
  float4 f0 = s[0], f1 = s[1];
  uint4 o;
  o.x = packbf2(f0.x, f0.y); o.y = packbf2(f0.z, f0.w);
  o.z = packbf2(f1.x, f1.y); o.w = packbf2(f1.z, f1.w);
  ((uint4*)xb)[i] = o;
}

// --------------------------- router + top-p --------------------------------
// one wave32 per token; lane 0 does E=8 softmax / sort / cumsum / top-p
__global__ __launch_bounds__(256) void router_kernel(
    const float* __restrict__ x, const float* __restrict__ rw,
    float* __restrict__ w) {
  const int lane = threadIdx.x & 31;
  const int wid  = threadIdx.x >> 5;
  const int t    = blockIdx.x * 8 + wid;

  const float* xr = x + (size_t)t * H_DIM;
  float acc[E_NUM] = {};
  for (int h = lane; h < H_DIM; h += 32) {
    float xv = xr[h];
#pragma unroll
    for (int e = 0; e < E_NUM; ++e) acc[e] += xv * rw[e * H_DIM + h];
  }
#pragma unroll
  for (int e = 0; e < E_NUM; ++e)
    for (int off = 16; off; off >>= 1) acc[e] += __shfl_xor(acc[e], off, 32);

  if (lane == 0) {
    float mx = acc[0];
#pragma unroll
    for (int e = 1; e < E_NUM; ++e) mx = fmaxf(mx, acc[e]);
    float p[E_NUM]; float sum = 0.f;
#pragma unroll
    for (int e = 0; e < E_NUM; ++e) { p[e] = __expf(acc[e] - mx); sum += p[e]; }
    float inv = __builtin_amdgcn_rcpf(sum);
#pragma unroll
    for (int e = 0; e < E_NUM; ++e) p[e] *= inv;

    int idx[E_NUM];
#pragma unroll
    for (int e = 0; e < E_NUM; ++e) idx[e] = e;
    // insertion sort, descending (mirrors argsort(-probs))
    for (int i = 1; i < E_NUM; ++i) {
      float pv = p[i]; int iv = idx[i]; int j = i - 1;
      while (j >= 0 && p[j] < pv) { p[j+1] = p[j]; idx[j+1] = idx[j]; --j; }
      p[j+1] = pv; idx[j+1] = iv;
    }
    float o8[E_NUM];
    float cum = 0.f; bool crossed = false;
    for (int i = 0; i < E_NUM; ++i) {
      cum += p[i];
      bool exceed = cum > TOPP;
      bool keep = !exceed || !crossed;   // keep while <= top_p, plus first cross
      if (exceed) crossed = true;
      o8[idx[i]] = keep ? p[i] : 0.f;
    }
#pragma unroll
    for (int e = 0; e < E_NUM; ++e) w[(size_t)t * E_NUM + e] = o8[e];
  }
}

// ------------------- fused gate+up GEMM, mid = silu(g)*u -------------------
// C[M=T, N=I] = x(bf16) @ W^T ; A and W^T are both K-contiguous row-major,
// so A and B WMMA fragments gather identically (two b128 ds loads per frag).
__global__ __launch_bounds__(256) void gateup_kernel(
    const unsigned short* __restrict__ xb,   // [T, H] bf16
    const float* __restrict__ gw,            // [I, H] f32
    const float* __restrict__ uw,            // [I, H] f32
    unsigned short* __restrict__ mid) {      // [T, I] bf16
  __shared__ __align__(16) unsigned short lA [2][BM * LDT];
  __shared__ __align__(16) unsigned short lBg[2][BN * LDT];
  __shared__ __align__(16) unsigned short lBu[2][BN * LDT];

  const int tid  = threadIdx.x;
  const int lane = tid & 31, wid = tid >> 5;
  const int wm = wid >> 1, wn = wid & 1;       // 4 x 2 waves
  const int m0 = blockIdx.y * BM;
  const int n0 = blockIdx.x * BN;
  const int hsel = lane >> 4, lr = lane & 15;  // lane half / row-in-half
  const int koff = hsel * 8;

  v8f accg[2][2] = {};
  v8f accu[2][2] = {};

  // A tile: 128x32 bf16 = 512 x 16B chunks, async global->LDS
  auto stageA = [&](int k0, int buf) {
#pragma unroll
    for (int i = 0; i < 2; ++i) {
      int id = tid + i * 256;                 // 0..511
      int row = id >> 2, kc = id & 3;
      async_ld_b128(lds_off32(&lA[buf][row * LDT + kc * 8]),
                    xb + (size_t)(m0 + row) * H_DIM + k0 + kc * 8);
    }
  };
  // B tiles (gate/up): global f32 -> regs
  auto loadB = [&](int k0, float4 fr[4]) {
#pragma unroll
    for (int i = 0; i < 2; ++i) {
      int id = tid + i * 256;                 // bit8 selects matrix
      const float* src = (id & 256) ? uw : gw;
      int rid = id & 255;
      int row = rid >> 2, kc = rid & 3;
      const float4* s =
          (const float4*)(src + (size_t)(n0 + row) * H_DIM + k0 + kc * 8);
      fr[2 * i] = s[0]; fr[2 * i + 1] = s[1];
    }
  };
  // regs -> bf16 -> LDS
  auto storeB = [&](int buf, const float4 fr[4]) {
#pragma unroll
    for (int i = 0; i < 2; ++i) {
      int id = tid + i * 256;
      unsigned short* dst = (id & 256) ? lBu[buf] : lBg[buf];
      int rid = id & 255;
      int row = rid >> 2, kc = rid & 3;
      uint4 v;
      v.x = packbf2(fr[2*i].x, fr[2*i].y);   v.y = packbf2(fr[2*i].z, fr[2*i].w);
      v.z = packbf2(fr[2*i+1].x, fr[2*i+1].y); v.w = packbf2(fr[2*i+1].z, fr[2*i+1].w);
      *(uint4*)&dst[row * LDT + kc * 8] = v;
    }
  };

  // prologue: stage tile 0 into buffer 0
  {
    float4 fr[4];
    stageA(0, 0);
    loadB(0, fr);
    storeB(0, fr);
  }
  wait_async0();
  __syncthreads();

  int cur = 0;
  for (int k0 = 0; k0 < H_DIM; k0 += BK) {
    const int nxt = cur ^ 1;
    const bool more = (k0 + BK) < H_DIM;
    float4 fr[4];
    if (more) {
      stageA(k0 + BK, nxt);     // async copies proceed under the WMMAs
      loadB(k0 + BK, fr);       // global loads in flight during compute
    }

    FragU a[2], bg[2], bu[2];
#pragma unroll
    for (int mi = 0; mi < 2; ++mi) {
      int r = (wm * 32 + mi * 16 + lr) * LDT;
      a[mi].q[0] = *(const uint4*)&lA[cur][r + koff];
      a[mi].q[1] = *(const uint4*)&lA[cur][r + 16 + koff];
    }
#pragma unroll
    for (int ni = 0; ni < 2; ++ni) {
      int r = (wn * 32 + ni * 16 + lr) * LDT;
      bg[ni].q[0] = *(const uint4*)&lBg[cur][r + koff];
      bg[ni].q[1] = *(const uint4*)&lBg[cur][r + 16 + koff];
      bu[ni].q[0] = *(const uint4*)&lBu[cur][r + koff];
      bu[ni].q[1] = *(const uint4*)&lBu[cur][r + 16 + koff];
    }
#pragma unroll
    for (int mi = 0; mi < 2; ++mi)
#pragma unroll
      for (int ni = 0; ni < 2; ++ni) {
        accg[mi][ni] = __builtin_amdgcn_wmma_f32_16x16x32_bf16(
            false, a[mi].f, false, bg[ni].f, (short)0, accg[mi][ni], false, false);
        accu[mi][ni] = __builtin_amdgcn_wmma_f32_16x16x32_bf16(
            false, a[mi].f, false, bu[ni].f, (short)0, accu[mi][ni], false, false);
      }

    if (more) storeB(nxt, fr);
    wait_async0();
    __syncthreads();
    cur = nxt;
  }

  // epilogue: mid = silu(g) * u  (bf16)
#pragma unroll
  for (int mi = 0; mi < 2; ++mi)
#pragma unroll
    for (int ni = 0; ni < 2; ++ni) {
      int n = n0 + wn * 32 + ni * 16 + lr;
#pragma unroll
      for (int v = 0; v < 8; ++v) {
        int m = m0 + wm * 32 + mi * 16 + hsel * 8 + v;
        float g = accg[mi][ni][v];
        float u = accu[mi][ni][v];
        float sg = g * __builtin_amdgcn_rcpf(1.0f + __expf(-g));   // silu
        mid[(size_t)m * I_DIM + n] = f2bf_rne(sg * u);
      }
    }
}

// ---------------- down GEMM + routed accumulate into out -------------------
__global__ __launch_bounds__(256) void down_kernel(
    const unsigned short* __restrict__ mid,  // [T, I] bf16
    const float* __restrict__ dw,            // [H, I] f32
    const float* __restrict__ wrt,           // [T, E] router weights
    float* __restrict__ out,                 // [T, H] f32
    int expert, int beta) {
  __shared__ __align__(16) unsigned short lA[2][BM * LDT];
  __shared__ __align__(16) unsigned short lB[2][BN * LDT];

  const int tid  = threadIdx.x;
  const int lane = tid & 31, wid = tid >> 5;
  const int wm = wid >> 1, wn = wid & 1;
  const int m0 = blockIdx.y * BM;
  const int n0 = blockIdx.x * BN;
  const int hsel = lane >> 4, lr = lane & 15;
  const int koff = hsel * 8;

  v8f acc[2][2] = {};

  auto stageA = [&](int k0, int buf) {
#pragma unroll
    for (int i = 0; i < 2; ++i) {
      int id = tid + i * 256;
      int row = id >> 2, kc = id & 3;
      async_ld_b128(lds_off32(&lA[buf][row * LDT + kc * 8]),
                    mid + (size_t)(m0 + row) * I_DIM + k0 + kc * 8);
    }
  };
  auto loadB = [&](int k0, float4 fr[2]) {
    int row = tid >> 2, kc = tid & 3;        // 64x32 f32 = exactly 256 chunks
    const float4* s =
        (const float4*)(dw + (size_t)(n0 + row) * I_DIM + k0 + kc * 8);
    fr[0] = s[0]; fr[1] = s[1];
  };
  auto storeB = [&](int buf, const float4 fr[2]) {
    int row = tid >> 2, kc = tid & 3;
    uint4 v;
    v.x = packbf2(fr[0].x, fr[0].y); v.y = packbf2(fr[0].z, fr[0].w);
    v.z = packbf2(fr[1].x, fr[1].y); v.w = packbf2(fr[1].z, fr[1].w);
    *(uint4*)&lB[buf][row * LDT + kc * 8] = v;
  };

  {
    float4 fr[2];
    stageA(0, 0);
    loadB(0, fr);
    storeB(0, fr);
  }
  wait_async0();
  __syncthreads();

  int cur = 0;
  for (int k0 = 0; k0 < I_DIM; k0 += BK) {
    const int nxt = cur ^ 1;
    const bool more = (k0 + BK) < I_DIM;
    float4 fr[2];
    if (more) {
      stageA(k0 + BK, nxt);
      loadB(k0 + BK, fr);
    }

    FragU a[2], b[2];
#pragma unroll
    for (int mi = 0; mi < 2; ++mi) {
      int r = (wm * 32 + mi * 16 + lr) * LDT;
      a[mi].q[0] = *(const uint4*)&lA[cur][r + koff];
      a[mi].q[1] = *(const uint4*)&lA[cur][r + 16 + koff];
    }
#pragma unroll
    for (int ni = 0; ni < 2; ++ni) {
      int r = (wn * 32 + ni * 16 + lr) * LDT;
      b[ni].q[0] = *(const uint4*)&lB[cur][r + koff];
      b[ni].q[1] = *(const uint4*)&lB[cur][r + 16 + koff];
    }
#pragma unroll
    for (int mi = 0; mi < 2; ++mi)
#pragma unroll
      for (int ni = 0; ni < 2; ++ni)
        acc[mi][ni] = __builtin_amdgcn_wmma_f32_16x16x32_bf16(
            false, a[mi].f, false, b[ni].f, (short)0, acc[mi][ni], false, false);

    if (more) storeB(nxt, fr);
    wait_async0();
    __syncthreads();
    cur = nxt;
  }

#pragma unroll
  for (int mi = 0; mi < 2; ++mi)
#pragma unroll
    for (int ni = 0; ni < 2; ++ni) {
      int n = n0 + wn * 32 + ni * 16 + lr;
#pragma unroll
      for (int v = 0; v < 8; ++v) {
        int m = m0 + wm * 32 + mi * 16 + hsel * 8 + v;
        float wgt = wrt[(size_t)m * E_NUM + expert];
        float val = acc[mi][ni][v] * wgt;
        size_t o = (size_t)m * H_DIM + n;
        out[o] = beta ? (out[o] + val) : val;   // expert 0 initializes d_out
      }
    }
}

// ----------------------------- launch --------------------------------------
extern "C" void kernel_launch(void* const* d_in, const int* in_sizes, int n_in,
                              void* d_out, int out_size, void* d_ws,
                              size_t ws_size, hipStream_t stream) {
  (void)in_sizes; (void)n_in; (void)out_size; (void)ws_size;
  const float* x  = (const float*)d_in[0];   // [S,B,H]  -> [T,H]
  const float* rw = (const float*)d_in[1];   // [E,H]
  const float* gw = (const float*)d_in[2];   // [E,I,H]
  const float* uw = (const float*)d_in[3];   // [E,I,H]
  const float* dw = (const float*)d_in[4];   // [E,H,I]
  float* out = (float*)d_out;

  unsigned short* xb = (unsigned short*)d_ws;                       // 32 MiB
  float* wrt = (float*)((char*)d_ws + (size_t)T_TOK * H_DIM * 2);   // 256 KiB
  unsigned short* mid = (unsigned short*)((char*)d_ws +
      (size_t)T_TOK * H_DIM * 2 + (size_t)T_TOK * E_NUM * 4);       // 64 MiB

  // x -> bf16 (reused by all 16 GEMMs)
  cvt_x_kernel<<<(unsigned)((size_t)T_TOK * H_DIM / 8 / 256), 256, 0, stream>>>(
      x, xb);
  // router softmax + top-p weights
  router_kernel<<<T_TOK / 8, 256, 0, stream>>>(x, rw, wrt);

  for (int e = 0; e < E_NUM; ++e) {
    gateup_kernel<<<dim3(I_DIM / BN, T_TOK / BM), 256, 0, stream>>>(
        xb, gw + (size_t)e * I_DIM * H_DIM, uw + (size_t)e * I_DIM * H_DIM,
        mid);
    down_kernel<<<dim3(H_DIM / BN, T_TOK / BM), 256, 0, stream>>>(
        mid, dw + (size_t)e * H_DIM * I_DIM, wrt, out, e, e != 0);
  }
}
